// Net_51513837748925
// MI455X (gfx1250) — matrix operand order repaired and checked by reading, not compiled, not run on hardware
//
#include <hip/hip_runtime.h>
#include <hip/hip_bf16.h>

// ---------------------------------------------------------------------------
// 2-layer GCN forward (GCNConv -> ReLU -> GCNConv -> log_softmax) for gfx1250.
// Dense GEMMs use v_wmma_f32_16x16x32_f16 (f16 in, f32 accumulate).
// Edge aggregation stays fp32 with native global_atomic_add_f32 (bandwidth-
// bound phase, ~50us at 23.3 TB/s HBM; GEMMs are ~3.5us of traffic).
// ---------------------------------------------------------------------------

#define N_NODES   50000
#define N_EDGES   800000
#define D_IN      256
#define HIDDEN    128
#define N_CLASSES 40
#define NC_PAD    48            // 3 tiles of 16 for GEMM2

typedef __attribute__((ext_vector_type(16))) _Float16 v16h;
typedef __attribute__((ext_vector_type(8)))  float    v8f;

// ---------------------------------------------------------------------------
// Degree / normalization
// ---------------------------------------------------------------------------
__global__ void init_deg_kernel(float* __restrict__ deg) {
    int i = blockIdx.x * blockDim.x + threadIdx.x;
    if (i < N_NODES) deg[i] = 1.0f;                    // self-loop
}

__global__ void count_deg_kernel(const int* __restrict__ dst, float* __restrict__ deg) {
    int e = blockIdx.x * blockDim.x + threadIdx.x;
    if (e < N_EDGES) atomicAdd(&deg[dst[e]], 1.0f);
}

__global__ void rsqrt_deg_kernel(float* __restrict__ dis) {
    int i = blockIdx.x * blockDim.x + threadIdx.x;
    if (i < N_NODES) dis[i] = rsqrtf(dis[i]);          // in-place deg -> deg^-1/2
}

// ---------------------------------------------------------------------------
// Weight conversion fp32 -> f16 (W2 zero-padded from 40 to 48 columns)
// ---------------------------------------------------------------------------
__global__ void cvt_w1_kernel(const float* __restrict__ W1, _Float16* __restrict__ W1h) {
    int i = blockIdx.x * blockDim.x + threadIdx.x;
    if (i < D_IN * HIDDEN) W1h[i] = (_Float16)W1[i];
}

__global__ void cvt_w2_kernel(const float* __restrict__ W2, _Float16* __restrict__ W2h) {
    int i = blockIdx.x * blockDim.x + threadIdx.x;
    if (i < HIDDEN * NC_PAD) {
        int r = i / NC_PAD, c = i % NC_PAD;
        W2h[i] = (c < N_CLASSES) ? (_Float16)W2[r * N_CLASSES + c] : (_Float16)0.0f;
    }
}

// ---------------------------------------------------------------------------
// GEMM1: XW1[50000,128] = X[50000,256] @ W1[256,128]  (f16 WMMA, f32 acc)
// One wave computes a 16x128 strip: A fragment loaded once per k-step,
// 8 WMMAs per k-step into 8 accumulators. X is read exactly once from HBM.
// ---------------------------------------------------------------------------
#define G1_TILES ((N_NODES + 15) / 16)   // 3125

__global__ void gemm1_wmma_kernel(const float* __restrict__ X,
                                  const _Float16* __restrict__ W1h,
                                  float* __restrict__ XW) {
    const int wave = (blockIdx.x * blockDim.x + threadIdx.x) >> 5;
    if (wave >= G1_TILES) return;                       // wave-uniform exit
    const int lane = threadIdx.x & 31;
    const int r    = lane & 15;                         // row / col within tile
    const int half = lane >> 4;

    const int m = wave * 16 + r;                        // exact: 3125*16 == 50000
    const float* __restrict__ xrow = X + (size_t)m * D_IN;

    v8f acc[8];
    #pragma unroll
    for (int t = 0; t < 8; ++t) acc[t] = (v8f)(0.0f);

    #pragma unroll
    for (int kk = 0; kk < D_IN / 32; ++kk) {            // 8 k-steps of 32
        const int k0 = kk * 32;
        // ---- A fragment: lane holds row r, K = {k0+half*8 .. +7, +16 .. +23}
        const float* ap = xrow + k0 + half * 8;
        float4 x0 = *(const float4*)(ap + 0);
        float4 x1 = *(const float4*)(ap + 4);
        float4 x2 = *(const float4*)(ap + 16);
        float4 x3 = *(const float4*)(ap + 20);
        v16h a;
        a[0]=(_Float16)x0.x;  a[1]=(_Float16)x0.y;  a[2]=(_Float16)x0.z;  a[3]=(_Float16)x0.w;
        a[4]=(_Float16)x1.x;  a[5]=(_Float16)x1.y;  a[6]=(_Float16)x1.z;  a[7]=(_Float16)x1.w;
        a[8]=(_Float16)x2.x;  a[9]=(_Float16)x2.y;  a[10]=(_Float16)x2.z; a[11]=(_Float16)x2.w;
        a[12]=(_Float16)x3.x; a[13]=(_Float16)x3.y; a[14]=(_Float16)x3.z; a[15]=(_Float16)x3.w;

        const int kb = k0 + half * 8;                   // B: lane holds col N=lane&15, K = {kb..+7, kb+16..+23}
        #pragma unroll
        for (int nt = 0; nt < 8; ++nt) {
            v16h b;
            #pragma unroll
            for (int j = 0; j < 8; ++j) {
                b[j]     = W1h[(size_t)(kb + j) * HIDDEN + nt * 16 + (lane & 15)];
                b[8 + j] = W1h[(size_t)(kb + 16 + j) * HIDDEN + nt * 16 + (lane & 15)];
            }
            acc[nt] = __builtin_amdgcn_wmma_f32_16x16x32_f16(
                false, a, false, b, (short)0, acc[nt], false, false);
        }
    }

    // ---- store: VGPR d -> M = d + half*8, N = lane&15
    #pragma unroll
    for (int nt = 0; nt < 8; ++nt) {
        #pragma unroll
        for (int d = 0; d < 8; ++d) {
            int row = wave * 16 + d + half * 8;
            XW[(size_t)row * HIDDEN + nt * 16 + (lane & 15)] = acc[nt][d];
        }
    }
}

// ---------------------------------------------------------------------------
// Layer-1 aggregation: agg1 init = self-loop term, then edge scatter atomics
// ---------------------------------------------------------------------------
__global__ void selfloop1_kernel(const float* __restrict__ XW,
                                 const float* __restrict__ dis,
                                 float* __restrict__ agg) {
    int idx = blockIdx.x * blockDim.x + threadIdx.x;
    if (idx < N_NODES * HIDDEN) {
        int i = idx >> 7;
        float w = dis[i];
        agg[idx] = XW[idx] * w * w;
    }
}

__global__ void scatter1_kernel(const int* __restrict__ src, const int* __restrict__ dst,
                                const float* __restrict__ XW, const float* __restrict__ dis,
                                float* __restrict__ agg) {
    int idx = blockIdx.x * blockDim.x + threadIdx.x;   // E * 32 threads (4 feats each)
    if (idx >= N_EDGES * 32) return;
    int e = idx >> 5;
    int f = (idx & 31) * 4;
    int s = src[e], d = dst[e];
    float nrm = dis[s] * dis[d];
    float4 v = *(const float4*)(XW + (size_t)s * HIDDEN + f);
    float* o = agg + (size_t)d * HIDDEN + f;
    atomicAdd(o + 0, v.x * nrm);
    atomicAdd(o + 1, v.y * nrm);
    atomicAdd(o + 2, v.z * nrm);
    atomicAdd(o + 3, v.w * nrm);
}

// ---------------------------------------------------------------------------
// H = relu(agg1 + b1), converted straight to f16 (only consumed by GEMM2)
// ---------------------------------------------------------------------------
__global__ void relu_bias_kernel(const float* __restrict__ agg,
                                 const float* __restrict__ b1,
                                 _Float16* __restrict__ Hh) {
    int idx = blockIdx.x * blockDim.x + threadIdx.x;
    if (idx < N_NODES * HIDDEN) {
        float v = agg[idx] + b1[idx & 127];
        Hh[idx] = (_Float16)fmaxf(v, 0.0f);
    }
}

// ---------------------------------------------------------------------------
// GEMM2: XW2[50000,40] = H[50000,128] @ W2[128,40]  (padded to 48 cols)
// ---------------------------------------------------------------------------
__global__ void gemm2_wmma_kernel(const _Float16* __restrict__ Hh,
                                  const _Float16* __restrict__ W2h,
                                  float* __restrict__ XW2) {
    const int wave = (blockIdx.x * blockDim.x + threadIdx.x) >> 5;
    if (wave >= G1_TILES) return;
    const int lane = threadIdx.x & 31;
    const int r    = lane & 15;
    const int half = lane >> 4;
    const int m = wave * 16 + r;
    const _Float16* __restrict__ hrow = Hh + (size_t)m * HIDDEN;

    v8f acc[3];
    #pragma unroll
    for (int t = 0; t < 3; ++t) acc[t] = (v8f)(0.0f);

    #pragma unroll
    for (int kk = 0; kk < HIDDEN / 32; ++kk) {          // 4 k-steps
        const int k0 = kk * 32;
        const _Float16* ap = hrow + k0 + half * 8;
        v16h a;
        #pragma unroll
        for (int j = 0; j < 8; ++j) { a[j] = ap[j]; a[8 + j] = ap[16 + j]; }

        const int kb = k0 + half * 8;
        #pragma unroll
        for (int nt = 0; nt < 3; ++nt) {
            v16h b;
            #pragma unroll
            for (int j = 0; j < 8; ++j) {
                b[j]     = W2h[(size_t)(kb + j) * NC_PAD + nt * 16 + (lane & 15)];
                b[8 + j] = W2h[(size_t)(kb + 16 + j) * NC_PAD + nt * 16 + (lane & 15)];
            }
            acc[nt] = __builtin_amdgcn_wmma_f32_16x16x32_f16(
                false, a, false, b, (short)0, acc[nt], false, false);
        }
    }

    #pragma unroll
    for (int nt = 0; nt < 3; ++nt) {
        int col = nt * 16 + (lane & 15);
        if (col < N_CLASSES) {
            #pragma unroll
            for (int d = 0; d < 8; ++d) {
                int row = wave * 16 + d + half * 8;
                XW2[(size_t)row * N_CLASSES + col] = acc[nt][d];
            }
        }
    }
}

// ---------------------------------------------------------------------------
// Layer-2 aggregation
// ---------------------------------------------------------------------------
__global__ void selfloop2_kernel(const float* __restrict__ XW2,
                                 const float* __restrict__ dis,
                                 float* __restrict__ agg) {
    int idx = blockIdx.x * blockDim.x + threadIdx.x;
    if (idx < N_NODES * N_CLASSES) {
        int i = idx / N_CLASSES;
        float w = dis[i];
        agg[idx] = XW2[idx] * w * w;
    }
}

__global__ void scatter2_kernel(const int* __restrict__ src, const int* __restrict__ dst,
                                const float* __restrict__ XW2, const float* __restrict__ dis,
                                float* __restrict__ agg) {
    long long idx = (long long)blockIdx.x * blockDim.x + threadIdx.x;   // E * 40 threads
    if (idx >= (long long)N_EDGES * N_CLASSES) return;
    int e = (int)(idx / N_CLASSES);
    int f = (int)(idx % N_CLASSES);
    int s = src[e], d = dst[e];
    float nrm = dis[s] * dis[d];
    atomicAdd(agg + (size_t)d * N_CLASSES + f, XW2[(size_t)s * N_CLASSES + f] * nrm);
}

// ---------------------------------------------------------------------------
// out = log_softmax(agg2 + b2)
// ---------------------------------------------------------------------------
__global__ void logsoftmax_kernel(const float* __restrict__ agg,
                                  const float* __restrict__ b2,
                                  float* __restrict__ out) {
    int i = blockIdx.x * blockDim.x + threadIdx.x;
    if (i >= N_NODES) return;
    const float* row = agg + (size_t)i * N_CLASSES;
    float v[N_CLASSES];
    float m = -3.4e38f;
    #pragma unroll
    for (int f = 0; f < N_CLASSES; ++f) { v[f] = row[f] + b2[f]; m = fmaxf(m, v[f]); }
    float s = 0.0f;
    #pragma unroll
    for (int f = 0; f < N_CLASSES; ++f) s += __expf(v[f] - m);
    float l = m + __logf(s);
    float* o = out + (size_t)i * N_CLASSES;
    #pragma unroll
    for (int f = 0; f < N_CLASSES; ++f) o[f] = v[f] - l;
}

// ---------------------------------------------------------------------------
// Launch
// ---------------------------------------------------------------------------
extern "C" void kernel_launch(void* const* d_in, const int* in_sizes, int n_in,
                              void* d_out, int out_size, void* d_ws, size_t ws_size,
                              hipStream_t stream) {
    (void)in_sizes; (void)n_in; (void)out_size; (void)ws_size;

    const float* X   = (const float*)d_in[0];
    const int*   EI  = (const int*)d_in[1];
    const int*   src = EI;
    const int*   dst = EI + N_EDGES;
    const float* W1  = (const float*)d_in[2];
    const float* b1  = (const float*)d_in[3];
    const float* W2  = (const float*)d_in[4];
    const float* b2  = (const float*)d_in[5];
    float*       out = (float*)d_out;

    // Workspace layout (floats), all 16B-aligned; total ~80.3 MB.
    float* ws   = (float*)d_ws;
    float* dis  = ws;                                   // 50000   (deg -> deg^-1/2)
    float* xw1  = dis  + N_NODES;                       // 6.4M
    float* agg1 = xw1  + (size_t)N_NODES * HIDDEN;      // 6.4M
    _Float16* w1h = (_Float16*)(agg1 + (size_t)N_NODES * HIDDEN);   // 32768 halves
    _Float16* w2h = w1h + D_IN * HIDDEN;                // 6144 halves
    _Float16* hh  = w2h + HIDDEN * NC_PAD;              // 6.4M halves
    float* xw2  = (float*)(hh + (size_t)N_NODES * HIDDEN);          // 2M
    float* agg2 = xw2 + (size_t)N_NODES * N_CLASSES;    // 2M

    const int B = 256;
    #define GRID(n) ((int)(((long long)(n) + B - 1) / B))

    // 1) symmetric normalization  dis = rsqrt(1 + in-degree)
    init_deg_kernel <<<GRID(N_NODES), B, 0, stream>>>(dis);
    count_deg_kernel<<<GRID(N_EDGES), B, 0, stream>>>(dst, dis);
    rsqrt_deg_kernel<<<GRID(N_NODES), B, 0, stream>>>(dis);

    // 2) weights -> f16 (L2-resident for the GEMMs)
    cvt_w1_kernel<<<GRID(D_IN * HIDDEN), B, 0, stream>>>(W1, w1h);
    cvt_w2_kernel<<<GRID(HIDDEN * NC_PAD), B, 0, stream>>>(W2, w2h);

    // 3) layer 1: XW1 = X @ W1 (WMMA), then normalized scatter-add, relu+bias
    gemm1_wmma_kernel<<<GRID(G1_TILES * 32), B, 0, stream>>>(X, w1h, xw1);
    selfloop1_kernel <<<GRID(N_NODES * HIDDEN), B, 0, stream>>>(xw1, dis, agg1);
    scatter1_kernel  <<<GRID((long long)N_EDGES * 32), B, 0, stream>>>(src, dst, xw1, dis, agg1);
    relu_bias_kernel <<<GRID(N_NODES * HIDDEN), B, 0, stream>>>(agg1, b1, hh);

    // 4) layer 2: XW2 = H @ W2 (WMMA), scatter-add, bias + log_softmax
    gemm2_wmma_kernel<<<GRID(G1_TILES * 32), B, 0, stream>>>(hh, w2h, xw2);
    selfloop2_kernel <<<GRID(N_NODES * N_CLASSES), B, 0, stream>>>(xw2, dis, agg2);
    scatter2_kernel  <<<GRID((long long)N_EDGES * N_CLASSES), B, 0, stream>>>(src, dst, xw2, dis, agg2);
    logsoftmax_kernel<<<GRID(N_NODES), B, 0, stream>>>(agg2, b2, out);

    #undef GRID
}
